// GCN4_37434934952293
// MI455X (gfx1250) — compile-verified, matrix operand order checked by reading, not run on hardware
//
#include <hip/hip_runtime.h>
#include <math.h>

// ---------------- constants ----------------
#define H 128
#define BM 128
#define BN 64
#define BK 32
#define KW 16            // BK/2 packed bf16 pairs
#define LDP 28           // LDS row stride in dwords (16B-aligned, conflict-free)
#define ACT_NONE 0
#define ACT_SILU 1

#define N_NODES 20000
#define N_EDGES 320000
#define NB 32
#define NE2 496
#define C2 300
#define C4 150
#define C6 100
#define NCLS 600

typedef __attribute__((ext_vector_type(16))) __bf16 v16bf;
typedef __attribute__((ext_vector_type(8)))  unsigned int v8u;
typedef __attribute__((ext_vector_type(8)))  float v8f;

__device__ __forceinline__ unsigned short f2bf(float f) {
  unsigned u = __float_as_uint(f);
  u += 0x7FFFu + ((u >> 16) & 1u);          // round-to-nearest-even
  return (unsigned short)(u >> 16);
}
__device__ __forceinline__ unsigned pack2(float a, float b) {
  return (unsigned)f2bf(a) | ((unsigned)f2bf(b) << 16);
}
__device__ __forceinline__ float siluf(float x) { return x / (1.f + __expf(-x)); }

// ---------------- weight fp32 -> bf16 (zero K-padded) ----------------
__global__ void eg_convw(const float* __restrict__ w, unsigned short* __restrict__ o,
                         int K, int N, int Kpad) {
  long long i = (long long)blockIdx.x * blockDim.x + threadIdx.x;
  if (i >= (long long)Kpad * N) return;
  int k = (int)(i / N), n = (int)(i % N);
  o[i] = (k < K) ? f2bf(w[(size_t)k * N + n]) : (unsigned short)0;
}

// ---------------- WMMA GEMM ----------------
// MODE 0: A = P0 [M, lda] (plain, row major)
// MODE 1: edge gather: k<H -> P0[row[e]*H+k]; k<2H -> P0[col[e]*H+k-H];
//         k==2H -> radial[e]; k==2H+1 -> P1[e] (edge_attr)
// MODE 2: node concat: k<H -> P0[m*H+k]; else P1[m*H+k-H]
template <int MODE>
__global__ __launch_bounds__(256) void eg_gemm(
    const float* __restrict__ P0, int lda, const float* __restrict__ P1,
    const int* __restrict__ rowi, const int* __restrict__ coli,
    const float* __restrict__ radial, const unsigned short* __restrict__ Bw,
    const float* __restrict__ bias, float* __restrict__ C, int M, int N, int K,
    int Kpad, int act, int accum) {
  __shared__ unsigned int As4[BM][LDP];   // packed pairs of bf16 along K
  __shared__ unsigned int BsT[BN][LDP];   // B transposed: [n][k-pairs]
  __shared__ int ridxs[BM];
  __shared__ int cidxs[BM];

  const int tid = threadIdx.x;
  const int lane = tid & 31;
  const int wave = tid >> 5;      // 0..7
  const int wm = wave >> 1;       // 0..3
  const int wn = wave & 1;        // 0..1
  const int brow = blockIdx.x * BM, bcol = blockIdx.y * BN;
  const int half = lane >> 4, r = lane & 15;
  const bool fullM = (brow + BM) <= M;
  const bool fullN = (bcol + BN) <= N;

  if (MODE == 1) {
    for (int i = tid; i < BM; i += 256) {
      int gm = brow + i;
      ridxs[i] = (gm < M) ? rowi[gm] : 0;
      cidxs[i] = (gm < M) ? coli[gm] : 0;
    }
    __syncthreads();
  }

  auto fetchA = [&](int gm, int rr, int gk) -> float {
    if (gk >= K) return 0.f;
    if (MODE == 0) return P0[(size_t)gm * lda + gk];
    if (MODE == 1) {
      if (gk < H) return P0[(size_t)ridxs[rr] * H + gk];
      if (gk < 2 * H) return P0[(size_t)cidxs[rr] * H + (gk - H)];
      if (gk == 2 * H) return radial[gm];
      return P1[gm];
    }
    return (gk < H) ? P0[(size_t)gm * H + gk] : P1[(size_t)gm * H + (gk - H)];
  };

  v8f acc[2][2] = {};

  for (int k0 = 0; k0 < Kpad; k0 += BK) {
    const bool fullK = (k0 + BK) <= K;

    if (MODE == 0 && k0 + BK < Kpad) {   // prefetch next A tile (global_prefetch_b8)
      int rr = tid & (BM - 1);
      int gm = brow + rr, nk = k0 + BK;
      if (gm < M && nk < K) __builtin_prefetch(&P0[(size_t)gm * lda + nk], 0, 0);
    }

    // ---- stage A (BM x BK as BM x KW packed dwords) ----
    if (fullM && fullK) {
      for (int p = tid; p < BM * KW; p += 256) {
        int rr = p >> 4, jc = p & 15;
        int gk = k0 + 2 * jc;
        float v0, v1;
        if (MODE == 0) {
          const float* ap = P0 + (size_t)(brow + rr) * lda + gk;
          v0 = ap[0]; v1 = ap[1];
        } else if (MODE == 1) {
          // fullK with K=257/258 implies gk+1 < 2H, so only the h gathers occur
          int gk1 = gk + 1;
          v0 = (gk < H) ? P0[(size_t)ridxs[rr] * H + gk]
                        : P0[(size_t)cidxs[rr] * H + (gk - H)];
          v1 = (gk1 < H) ? P0[(size_t)ridxs[rr] * H + gk1]
                         : P0[(size_t)cidxs[rr] * H + (gk1 - H)];
        } else {
          int gm = brow + rr, gk1 = gk + 1;
          v0 = (gk < H) ? P0[(size_t)gm * H + gk] : P1[(size_t)gm * H + (gk - H)];
          v1 = (gk1 < H) ? P0[(size_t)gm * H + gk1] : P1[(size_t)gm * H + (gk1 - H)];
        }
        As4[rr][jc] = pack2(v0, v1);
      }
    } else {
      for (int p = tid; p < BM * KW; p += 256) {
        int rr = p >> 4, jc = p & 15;
        int gm = brow + rr, gk = k0 + 2 * jc;
        float v0 = 0.f, v1 = 0.f;
        if (gm < M) {
          v0 = fetchA(gm, rr, gk);
          v1 = fetchA(gm, rr, gk + 1);
        }
        As4[rr][jc] = pack2(v0, v1);
      }
    }

    // ---- stage B transposed (BN x KW packed dwords); Bw is zero-padded to Kpad ----
    if (fullN) {
      for (int p = tid; p < BN * KW; p += 256) {
        int jc = p >> 6, nc = p & 63;
        size_t o = (size_t)(k0 + 2 * jc) * N + (bcol + nc);
        BsT[nc][jc] = (unsigned)Bw[o] | ((unsigned)Bw[o + N] << 16);
      }
    } else {
      for (int p = tid; p < BN * KW; p += 256) {
        int jc = p >> 6, nc = p & 63;
        int gn = bcol + nc;
        unsigned lo = 0, hi = 0;
        if (gn < N) {
          size_t o = (size_t)(k0 + 2 * jc) * N + gn;
          lo = Bw[o]; hi = Bw[o + N];
        }
        BsT[nc][jc] = lo | (hi << 16);
      }
    }
    __syncthreads();

    // ---- fragments: two aligned 16B LDS loads each ----
    v8u au[2], bu[2];
#pragma unroll
    for (int f = 0; f < 2; ++f) {
      const unsigned int* ap = &As4[wm * 32 + f * 16 + r][half * 4];
#pragma unroll
      for (int j = 0; j < 4; ++j) { au[f][j] = ap[j]; au[f][4 + j] = ap[8 + j]; }
    }
#pragma unroll
    for (int f = 0; f < 2; ++f) {
      const unsigned int* bp = &BsT[wn * 32 + f * 16 + r][half * 4];
#pragma unroll
      for (int j = 0; j < 4; ++j) { bu[f][j] = bp[j]; bu[f][4 + j] = bp[8 + j]; }
    }
#pragma unroll
    for (int fm = 0; fm < 2; ++fm)
#pragma unroll
      for (int fn = 0; fn < 2; ++fn)
        acc[fm][fn] = __builtin_amdgcn_wmma_f32_16x16x32_bf16(
            false, __builtin_bit_cast(v16bf, au[fm]), false,
            __builtin_bit_cast(v16bf, bu[fn]), (short)0, acc[fm][fn], false,
            false);
    __syncthreads();
  }

  // ---- epilogue: C layout lane0-15 = N, vgpr v = M (+8 for upper half) ----
#pragma unroll
  for (int fm = 0; fm < 2; ++fm) {
#pragma unroll
    for (int fn = 0; fn < 2; ++fn) {
      int col = bcol + wn * 32 + fn * 16 + r;
      if (col >= N) continue;
#pragma unroll
      for (int v = 0; v < 8; ++v) {
        int row = brow + wm * 32 + fm * 16 + half * 8 + v;
        if (row >= M) continue;
        float x = acc[fm][fn][v] + bias[col];
        if (act == ACT_SILU) x = siluf(x);
        size_t o = (size_t)row * N + col;
        if (accum) x += C[o];
        C[o] = x;
      }
    }
  }
}

// ---------------- elementwise / scatter kernels ----------------
__global__ void eg_zero(float* p, int n) {
  int i = blockIdx.x * blockDim.x + threadIdx.x;
  if (i < n) p[i] = 0.f;
}
__global__ void eg_copy(float* dst, const float* src, int n) {
  int i = blockIdx.x * blockDim.x + threadIdx.x;
  if (i < n) dst[i] = src[i];
}
__global__ void eg_count(const int* __restrict__ ids, float* cnt, int n) {
  int i = blockIdx.x * blockDim.x + threadIdx.x;
  if (i < n) atomicAdd(&cnt[ids[i]], 1.f);
}
__global__ void eg_radialk(const float* __restrict__ pos, const int* __restrict__ rowi,
                           const int* __restrict__ coli, float* __restrict__ radial,
                           float* __restrict__ cdiff, int E) {
  int e = blockIdx.x * blockDim.x + threadIdx.x;
  if (e >= E) return;
  int a = rowi[e], b = coli[e];
  float dx = pos[a * 3] - pos[b * 3];
  float dy = pos[a * 3 + 1] - pos[b * 3 + 1];
  float dz = pos[a * 3 + 2] - pos[b * 3 + 2];
  float rad = dx * dx + dy * dy + dz * dz;
  float inv = 1.f / (sqrtf(rad) + 1.f);
  radial[e] = rad;
  cdiff[(size_t)e * 3]     = dx * inv;
  cdiff[(size_t)e * 3 + 1] = dy * inv;
  cdiff[(size_t)e * 3 + 2] = dz * inv;
}
// one wave per edge: m *= sigmoid(m . w + b)
__global__ void eg_att(float* __restrict__ m, const float* __restrict__ w,
                       const float* __restrict__ b, int E) {
  int e = (blockIdx.x * blockDim.x + threadIdx.x) >> 5;
  int lane = threadIdx.x & 31;
  if (e >= E) return;
  size_t base = (size_t)e * H;
  float v0 = m[base + lane], v1 = m[base + 32 + lane];
  float v2 = m[base + 64 + lane], v3 = m[base + 96 + lane];
  float s = v0 * w[lane] + v1 * w[32 + lane] + v2 * w[64 + lane] + v3 * w[96 + lane];
  for (int off = 16; off; off >>= 1) s += __shfl_xor(s, off, 32);
  float g = 1.f / (1.f + __expf(-(s + b[0])));
  m[base + lane] = v0 * g;
  m[base + 32 + lane] = v1 * g;
  m[base + 64 + lane] = v2 * g;
  m[base + 96 + lane] = v3 * g;
}
// one wave per edge: t = tanh(q . w2); csum[row] += cdiff*t
__global__ void eg_coordt(const float* __restrict__ q, const float* __restrict__ w2,
                          const float* __restrict__ cdiff, const int* __restrict__ rowi,
                          float* __restrict__ csum, int E) {
  int e = (blockIdx.x * blockDim.x + threadIdx.x) >> 5;
  int lane = threadIdx.x & 31;
  if (e >= E) return;
  size_t base = (size_t)e * H;
  float s = q[base + lane] * w2[lane] + q[base + 32 + lane] * w2[32 + lane] +
            q[base + 64 + lane] * w2[64 + lane] + q[base + 96 + lane] * w2[96 + lane];
  for (int off = 16; off; off >>= 1) s += __shfl_xor(s, off, 32);
  float t = tanhf(s);
  if (lane < 3)
    atomicAdd(&csum[(size_t)rowi[e] * 3 + lane], cdiff[(size_t)e * 3 + lane] * t);
}
__global__ void eg_posupd(float* pos, const float* __restrict__ csum,
                          const float* __restrict__ cnt, int total) {
  int i = blockIdx.x * blockDim.x + threadIdx.x;
  if (i < total) pos[i] += csum[i] / fmaxf(cnt[i / 3], 1.f);
}
__global__ void eg_divcnt(float* x, const float* __restrict__ cnt, int C, int total) {
  int i = blockIdx.x * blockDim.x + threadIdx.x;
  if (i < total) x[i] /= fmaxf(cnt[i / C], 1.f);
}
__global__ void eg_agg(const float* __restrict__ m, const int* __restrict__ rowi,
                       float* __restrict__ agg, long long total) {
  long long i = (long long)blockIdx.x * blockDim.x + threadIdx.x;
  if (i >= total) return;
  int e = (int)(i >> 7), f = (int)(i & 127);
  atomicAdd(&agg[(size_t)rowi[e] * H + f], m[i]);
}
__global__ void eg_pool(const float* __restrict__ x, const int* __restrict__ ids,
                        float* __restrict__ pool, int n, int C) {
  long long i = (long long)blockIdx.x * blockDim.x + threadIdx.x;
  if (i >= (long long)n * C) return;
  int nn = (int)(i / C), cc = (int)(i % C);
  atomicAdd(&pool[(size_t)ids[nn] * C + cc], x[i]);
}
__global__ void eg_bnrelu(const float* __restrict__ x, const float* __restrict__ cnt,
                          const float* __restrict__ g, const float* __restrict__ be,
                          const float* __restrict__ mm, const float* __restrict__ vv,
                          float* __restrict__ out, int C, int ldout, int colofs,
                          int rows) {
  int i = blockIdx.x * blockDim.x + threadIdx.x;
  if (i >= rows * C) return;
  int rr = i / C, cc = i % C;
  float val = x[i];
  if (cnt) val /= fmaxf(cnt[rr], 1.f);
  val = g[cc] * (val - mm[cc]) * rsqrtf(vv[cc] + 1e-5f) + be[cc];
  out[(size_t)rr * ldout + colofs + cc] = fmaxf(val, 0.f);
}
__global__ void eg_bnact(float* x, const float* __restrict__ g,
                         const float* __restrict__ be, const float* __restrict__ mm,
                         const float* __restrict__ vv, int C, int rows, int dosig) {
  int i = blockIdx.x * blockDim.x + threadIdx.x;
  if (i >= rows * C) return;
  int cc = i % C;
  float val = x[i];
  val = g[cc] * (val - mm[cc]) * rsqrtf(vv[cc] + 1e-5f) + be[cc];
  val = fmaxf(val, 0.f);
  if (dosig) val = 1.f / (1.f + __expf(-val));
  x[i] = val;
}

// ---------------- host-side param unpack (jax tree_flatten order: sorted keys) --
struct Lin { const float *b, *w; };
struct BNP { const float *be, *g, *m, *v; };
struct Layer { Lin att, coord1; const float* coord2_w; Lin edge1, edge2, node1, node2; };
struct Egnn { Lin emb_in, emb_out; Layer layer[2]; };

static Lin takeLin(void* const* d, int& c) {
  Lin l; l.b = (const float*)d[c++]; l.w = (const float*)d[c++]; return l;
}
static BNP takeBN(void* const* d, int& c) {
  BNP b; b.be = (const float*)d[c++]; b.g = (const float*)d[c++];
  b.m = (const float*)d[c++]; b.v = (const float*)d[c++]; return b;
}
static Layer takeLayer(void* const* d, int& c) {
  Layer L;
  L.att = takeLin(d, c);
  L.coord1 = takeLin(d, c);
  L.coord2_w = (const float*)d[c++];
  L.edge1 = takeLin(d, c);
  L.edge2 = takeLin(d, c);
  L.node1 = takeLin(d, c);
  L.node2 = takeLin(d, c);
  return L;
}
static Egnn takeEgnn(void* const* d, int& c) {
  Egnn e;
  e.emb_in = takeLin(d, c);
  e.emb_out = takeLin(d, c);
  e.layer[0] = takeLayer(d, c);
  e.layer[1] = takeLayer(d, c);
  return e;
}

// ---------------- launch helpers ----------------
static inline void launch_gemm(int mode, const float* p0, int lda, const float* p1,
                               const int* rowi, const int* coli, const float* radial,
                               const float* W, int K, int N, const float* bias,
                               float* Cc, int M, int act, int accum,
                               unsigned short* wbuf, hipStream_t s) {
  int Kpad = (K + 31) & ~31;
  long long tot = (long long)Kpad * N;
  eg_convw<<<dim3((unsigned)((tot + 255) / 256)), dim3(256), 0, s>>>(W, wbuf, K, N, Kpad);
  dim3 grid((M + BM - 1) / BM, (N + BN - 1) / BN), blk(256);
  if (mode == 0)
    eg_gemm<0><<<grid, blk, 0, s>>>(p0, lda, p1, rowi, coli, radial, wbuf, bias, Cc, M, N, K, Kpad, act, accum);
  else if (mode == 1)
    eg_gemm<1><<<grid, blk, 0, s>>>(p0, lda, p1, rowi, coli, radial, wbuf, bias, Cc, M, N, K, Kpad, act, accum);
  else
    eg_gemm<2><<<grid, blk, 0, s>>>(p0, lda, p1, rowi, coli, radial, wbuf, bias, Cc, M, N, K, Kpad, act, accum);
}
static inline void zero_f(float* p, int n, hipStream_t s) {
  eg_zero<<<dim3((n + 255) / 256), dim3(256), 0, s>>>(p, n);
}

struct Bufs { float *h, *ntmp, *agg, *m1, *m, *rad, *cdf, *csum; };

static void egnn_run(const Egnn& P, const float* xin, int in_dim, float* hout,
                     int out_dim, float* pos, const int* rowi, const int* coli,
                     const float* eattr, int E, int NNn, const float* cnt,
                     const Bufs& B, unsigned short* wbuf, hipStream_t s) {
  launch_gemm(0, xin, in_dim, nullptr, nullptr, nullptr, nullptr, P.emb_in.w,
              in_dim, H, P.emb_in.b, B.h, NNn, ACT_NONE, 0, wbuf, s);
  for (int l = 0; l < 2; ++l) {
    const Layer& L = P.layer[l];
    eg_radialk<<<dim3((E + 255) / 256), dim3(256), 0, s>>>(pos, rowi, coli, B.rad, B.cdf, E);
    int Ke = eattr ? (2 * H + 2) : (2 * H + 1);
    launch_gemm(1, B.h, H, eattr, rowi, coli, B.rad, L.edge1.w, Ke, H, L.edge1.b,
                B.m1, E, ACT_SILU, 0, wbuf, s);
    launch_gemm(0, B.m1, H, nullptr, nullptr, nullptr, nullptr, L.edge2.w, H, H,
                L.edge2.b, B.m, E, ACT_SILU, 0, wbuf, s);
    eg_att<<<dim3((E + 7) / 8), dim3(256), 0, s>>>(B.m, L.att.w, L.att.b, E);
    // coord model
    launch_gemm(0, B.m, H, nullptr, nullptr, nullptr, nullptr, L.coord1.w, H, H,
                L.coord1.b, B.m1, E, ACT_SILU, 0, wbuf, s);
    zero_f(B.csum, NNn * 3, s);
    eg_coordt<<<dim3((E + 7) / 8), dim3(256), 0, s>>>(B.m1, L.coord2_w, B.cdf, rowi, B.csum, E);
    eg_posupd<<<dim3((NNn * 3 + 255) / 256), dim3(256), 0, s>>>(pos, B.csum, cnt, NNn * 3);
    // node model
    zero_f(B.agg, NNn * H, s);
    long long tot = (long long)E * H;
    eg_agg<<<dim3((unsigned)((tot + 255) / 256)), dim3(256), 0, s>>>(B.m, rowi, B.agg, tot);
    launch_gemm(2, B.h, H, B.agg, nullptr, nullptr, nullptr, L.node1.w, 2 * H, H,
                L.node1.b, B.ntmp, NNn, ACT_SILU, 0, wbuf, s);
    launch_gemm(0, B.ntmp, H, nullptr, nullptr, nullptr, nullptr, L.node2.w, H, H,
                L.node2.b, B.h, NNn, ACT_NONE, 1, wbuf, s);
  }
  launch_gemm(0, B.h, H, nullptr, nullptr, nullptr, nullptr, P.emb_out.w, H,
              out_dim, P.emb_out.b, hout, NNn, ACT_NONE, 0, wbuf, s);
}

// ---------------- entry point ----------------
extern "C" void kernel_launch(void* const* d_in, const int* in_sizes, int n_in,
                              void* d_out, int out_size, void* d_ws, size_t ws_size,
                              hipStream_t stream) {
  (void)in_sizes; (void)n_in; (void)out_size; (void)ws_size;
  int c = 0;
  const float* edge_attr = (const float*)d_in[c++];
  const int* edge_index = (const int*)d_in[c++];
  const int* edge_index2 = (const int*)d_in[c++];
  BNP bnr1 = takeBN(d_in, c), bnr2 = takeBN(d_in, c), bnr3 = takeBN(d_in, c);
  Egnn e1 = takeEgnn(d_in, c), e2 = takeEgnn(d_in, c), e3 = takeEgnn(d_in, c);
  Egnn e4 = takeEgnn(d_in, c), e5 = takeEgnn(d_in, c), e6 = takeEgnn(d_in, c);
  Lin fc1; fc1.b = (const float*)d_in[c++];
  BNP fc1bn = takeBN(d_in, c);
  fc1.w = (const float*)d_in[c++];
  Lin fin; fin.b = (const float*)d_in[c++];
  BNP finbn = takeBN(d_in, c);
  fin.w = (const float*)d_in[c++];
  const int* x_batch = (const int*)d_in[c++];
  const float* x_emb_seq = (const float*)d_in[c++];
  const float* x_pos = (const float*)d_in[c++];
  const float* x_res = (const float*)d_in[c++];

  const int* rowB = edge_index;            // [E]
  const int* colB = edge_index + N_EDGES;  // [E]
  const int* row2 = edge_index2;           // [496]
  const int* col2 = edge_index2 + NE2;

  // workspace bump allocator
  char* base = (char*)d_ws;
  size_t off = 0;
  auto alloc = [&](size_t elems, size_t esz) -> void* {
    void* p = base + off;
    off += (elems * esz + 255) & ~(size_t)255;
    return p;
  };
  unsigned short* wbuf = (unsigned short*)alloc((size_t)1120 * 800, 2);
  Bufs G;
  G.h = (float*)alloc((size_t)N_NODES * H, 4);
  G.ntmp = (float*)alloc((size_t)N_NODES * H, 4);
  G.agg = (float*)alloc((size_t)N_NODES * H, 4);
  G.m1 = (float*)alloc((size_t)N_EDGES * H, 4);
  G.m = (float*)alloc((size_t)N_EDGES * H, 4);
  G.rad = (float*)alloc(N_EDGES, 4);
  G.cdf = (float*)alloc((size_t)N_EDGES * 3, 4);
  G.csum = (float*)alloc((size_t)N_NODES * 3, 4);
  float* pos = (float*)alloc((size_t)N_NODES * 3, 4);
  float* cntN = (float*)alloc(N_NODES, 4);
  float* h1 = (float*)alloc((size_t)N_NODES * C2, 4);
  float* h2 = (float*)alloc((size_t)N_NODES * C4, 4);
  float* h3 = (float*)alloc((size_t)N_NODES * C6, 4);
  Bufs S;
  S.h = (float*)alloc((size_t)NB * H, 4);
  S.ntmp = (float*)alloc((size_t)NB * H, 4);
  S.agg = (float*)alloc((size_t)NB * H, 4);
  S.m1 = (float*)alloc((size_t)NE2 * H, 4);
  S.m = (float*)alloc((size_t)NE2 * H, 4);
  S.rad = (float*)alloc(NE2, 4);
  S.cdf = (float*)alloc((size_t)NE2 * 3, 4);
  S.csum = (float*)alloc((size_t)NB * 3, 4);
  float* poss = (float*)alloc((size_t)NB * 3, 4);
  float* cntB = (float*)alloc(NB, 4);
  float* h4 = (float*)alloc((size_t)NB * C2, 4);
  float* h5 = (float*)alloc((size_t)NB * C4, 4);
  float* h6 = (float*)alloc((size_t)NB * C6, 4);
  float* bcnt = (float*)alloc(NB, 4);
  float* pool = (float*)alloc((size_t)NB * C2, 4);
  float* ccat = (float*)alloc((size_t)NB * 1100, 4);
  float* fco = (float*)alloc((size_t)NB * 800, 4);
  float* outp = (float*)d_out;  // [32, 600]

  // ---- static per-graph counts (recomputed each call; deterministic) ----
  zero_f(cntN, N_NODES, stream);
  eg_count<<<dim3((N_EDGES + 255) / 256), dim3(256), 0, stream>>>(rowB, cntN, N_EDGES);
  zero_f(cntB, NB, stream);
  eg_count<<<dim3((NE2 + 255) / 256), dim3(256), 0, stream>>>(row2, cntB, NE2);
  zero_f(bcnt, NB, stream);
  eg_count<<<dim3((N_NODES + 255) / 256), dim3(256), 0, stream>>>(x_batch, bcnt, N_NODES);
  eg_copy<<<dim3((N_NODES * 3 + 255) / 256), dim3(256), 0, stream>>>(pos, x_pos, N_NODES * 3);

  // ---- big-graph EGNN chain ----
  egnn_run(e1, x_res, 1280, h1, C2, pos, rowB, colB, edge_attr, N_EDGES, N_NODES, cntN, G, wbuf, stream);
  egnn_run(e2, h1, C2, h2, C4, pos, rowB, colB, edge_attr, N_EDGES, N_NODES, cntN, G, wbuf, stream);
  egnn_run(e3, h2, C4, h3, C6, pos, rowB, colB, edge_attr, N_EDGES, N_NODES, cntN, G, wbuf, stream);

  // ---- batch-level EGNN chain ----
  zero_f(poss, NB * 3, stream);
  eg_pool<<<dim3((unsigned)(((long long)N_NODES * 3 + 255) / 256)), dim3(256), 0, stream>>>(
      x_pos, x_batch, poss, N_NODES, 3);
  eg_divcnt<<<dim3((NB * 3 + 255) / 256), dim3(256), 0, stream>>>(poss, bcnt, 3, NB * 3);
  egnn_run(e4, x_emb_seq, 1280, h4, C2, poss, row2, col2, nullptr, NE2, NB, cntB, S, wbuf, stream);
  egnn_run(e5, h4, C2, h5, C4, poss, row2, col2, nullptr, NE2, NB, cntB, S, wbuf, stream);
  egnn_run(e6, h5, C4, h6, C6, poss, row2, col2, nullptr, NE2, NB, cntB, S, wbuf, stream);

  // ---- pooled BN+ReLU into concat [32, 1100]: [o1,o4,o2,o5,o3,o6] ----
  struct { const float* h; BNP bn; int C; int ofs; } pooled[3] = {
      {h1, bnr1, C2, 0}, {h2, bnr2, C4, 600}, {h3, bnr3, C6, 900}};
  for (int i = 0; i < 3; ++i) {
    int Cc = pooled[i].C;
    zero_f(pool, NB * Cc, stream);
    long long tot = (long long)N_NODES * Cc;
    eg_pool<<<dim3((unsigned)((tot + 255) / 256)), dim3(256), 0, stream>>>(
        pooled[i].h, x_batch, pool, N_NODES, Cc);
    eg_bnrelu<<<dim3((NB * Cc + 255) / 256), dim3(256), 0, stream>>>(
        pool, bcnt, pooled[i].bn.g, pooled[i].bn.be, pooled[i].bn.m, pooled[i].bn.v,
        ccat, Cc, 1100, pooled[i].ofs, NB);
  }
  eg_bnrelu<<<dim3((NB * C2 + 255) / 256), dim3(256), 0, stream>>>(
      h4, nullptr, bnr1.g, bnr1.be, bnr1.m, bnr1.v, ccat, C2, 1100, 300, NB);
  eg_bnrelu<<<dim3((NB * C4 + 255) / 256), dim3(256), 0, stream>>>(
      h5, nullptr, bnr2.g, bnr2.be, bnr2.m, bnr2.v, ccat, C4, 1100, 750, NB);
  eg_bnrelu<<<dim3((NB * C6 + 255) / 256), dim3(256), 0, stream>>>(
      h6, nullptr, bnr3.g, bnr3.be, bnr3.m, bnr3.v, ccat, C6, 1100, 1000, NB);

  // ---- FC head ----
  launch_gemm(0, ccat, 1100, nullptr, nullptr, nullptr, nullptr, fc1.w, 1100, 800,
              fc1.b, fco, NB, ACT_NONE, 0, wbuf, stream);
  eg_bnact<<<dim3((NB * 800 + 255) / 256), dim3(256), 0, stream>>>(
      fco, fc1bn.g, fc1bn.be, fc1bn.m, fc1bn.v, 800, NB, 0);
  launch_gemm(0, fco, 800, nullptr, nullptr, nullptr, nullptr, fin.w, 800, NCLS,
              fin.b, outp, NB, ACT_NONE, 0, wbuf, stream);
  eg_bnact<<<dim3((NB * NCLS + 255) / 256), dim3(256), 0, stream>>>(
      outp, finbn.g, finbn.be, finbn.m, finbn.v, NCLS, NB, 1);
}
